// AttentionForCausalLM_17042430231394
// MI455X (gfx1250) — compile-verified, matrix operand order checked by reading, not run on hardware
//
#include <hip/hip_runtime.h>
#include <hip/hip_bf16.h>

// Causal attention forward (flash-attention-2), bf16 WMMA, fp32 accumulate.
// B=2, S=2048, H=32, D=128, start_pos=0.
// Pass 1: one-shot fp32->bf16 convert (V also transposed to (B,H,D,S)).
// Pass 2: FA2 with double-buffered GLOBAL_LOAD_ASYNC_TO_LDS_B128 K/V staging.

#define BB 2
#define SS 2048
#define HH 32
#define DD 128
#define QT 128   // q rows per block = 8 waves * 16
#define KT 64    // kv rows per tile
#define NW 8     // waves per block
#define KSTR 136 // LDS row stride (bf16) for K tile: 272B = 17*16 (b128-aligned)
#define VSTR 72  // LDS row stride (bf16) for V^T tile: 144B = 9*16
#define PSTR 72  // LDS row stride (bf16) for P staging
#define SMEM_BYTES ((2*KT*KSTR + 2*DD*VSTR + NW*16*PSTR) * 2)  // 90112

typedef __attribute__((ext_vector_type(16))) __bf16 v16bf;
typedef __attribute__((ext_vector_type(8)))  float  v8f;

__device__ __forceinline__ unsigned short bfbits(float f) {
  unsigned u = __builtin_bit_cast(unsigned, f);
  return (unsigned short)((u + 0x7fffu + ((u >> 16) & 1u)) >> 16);
}
__device__ __forceinline__ __bf16 f2bf(float f) {
  unsigned short s = bfbits(f);
  return __builtin_bit_cast(__bf16, s);
}

// WMMA 16-bit fragment element -> K index (A: lane=M, B: lane=N).
// e(0..15): v=e>>1, hh=e&1; lanes0-15: K={0..7,16..23}; lanes16-31: +8
__device__ __forceinline__ int frag_k(int e, int hi) {
  int v = e >> 1, hh = e & 1;
  return (v < 4 ? v * 2 + hh : 16 + (v - 4) * 2 + hh) + (hi ? 8 : 0);
}

// one per-lane async DMA: 16B global -> LDS, tracked by ASYNCcnt
__device__ __forceinline__ void async_b128(unsigned lds_off, const __bf16* ga) {
  asm volatile("global_load_async_to_lds_b128 %0, %1, off"
               :: "v"(lds_off), "v"(ga) : "memory");
}

// ---------------- pass 1: conversion kernels ----------------
__global__ __launch_bounds__(256) void cvt_bf16_kernel(
    const float* __restrict__ src, __bf16* __restrict__ dst, int n4) {
  int i = blockIdx.x * blockDim.x + threadIdx.x;
  if (i >= n4) return;
  float4 f = ((const float4*)src)[i];
  uint2 p;
  p.x = (unsigned)bfbits(f.x) | ((unsigned)bfbits(f.y) << 16);
  p.y = (unsigned)bfbits(f.z) | ((unsigned)bfbits(f.w) << 16);
  ((uint2*)dst)[i] = p;
}

// (B,S,H,D) fp32 -> (B,H,D,S) bf16 (coalesced stores along S)
__global__ __launch_bounds__(256) void cvtT_bf16_kernel(
    const float* __restrict__ src, __bf16* __restrict__ dst) {
  int i = blockIdx.x * blockDim.x + threadIdx.x;   // one per 4 outputs
  int s4  = i % (SS / 4);
  int rem = i / (SS / 4);
  int d = rem % DD;
  int h = (rem / DD) % HH;
  int b = rem / (DD * HH);
  int s = s4 * 4;
  unsigned short q[4];
  #pragma unroll
  for (int j = 0; j < 4; ++j)
    q[j] = bfbits(src[(((size_t)b * SS + s + j) * HH + h) * DD + d]);
  uint2 p;
  p.x = (unsigned)q[0] | ((unsigned)q[1] << 16);
  p.y = (unsigned)q[2] | ((unsigned)q[3] << 16);
  *(uint2*)(dst + (((size_t)b * HH + h) * DD + d) * SS + s) = p;
}

// ---------------- pass 2: flash attention on bf16 ----------------
__global__ __launch_bounds__(256, 1)
void fa_fwd_bf16_kernel(const __bf16* __restrict__ wq, const __bf16* __restrict__ wk,
                        const __bf16* __restrict__ wvT, float* __restrict__ out) {
  extern __shared__ __align__(16) char smem[];
  __bf16* kbuf = (__bf16*)smem;                 // 2 x [KT][KSTR]
  __bf16* vbuf = kbuf + 2 * KT * KSTR;          // 2 x [DD][VSTR]  (V^T: [d][kv])
  __bf16* pbuf = vbuf + 2 * DD * VSTR;          // NW x [16][PSTR]

  const int tid  = threadIdx.x;
  const int wave = tid >> 5;
  const int lane = tid & 31;
  const int lh   = lane & 15;
  const int hi   = (lane >> 4) & 1;

  const int qtiles = SS / QT;
  const int qt = blockIdx.x % qtiles;
  const int h  = (blockIdx.x / qtiles) % HH;
  const int b  = blockIdx.x / (qtiles * HH);

  const int qbase = qt * QT + wave * 16;
  const float scale = 0.08838834764831845f;     // 1/sqrt(128)

  const __bf16* kbase  = wk  + (((size_t)b * SS) * HH + h) * DD;      // +row*HH*DD
  const __bf16* vTbase = wvT + (((size_t)b * HH + h) * DD) * SS;      // +d*SS

  // ---- Q fragment (A-layout bf16, 4 chunks of K=32) ----
  v16bf aq[4];
  {
    const __bf16* qrow = wq + (((size_t)b * SS + (qbase + lh)) * HH + h) * DD;
    #pragma unroll
    for (int c = 0; c < 4; ++c)
      #pragma unroll
      for (int e = 0; e < 16; ++e)
        aq[c][e] = qrow[c * 32 + frag_k(e, hi)];
  }

  v8f acc[8];
  #pragma unroll
  for (int i = 0; i < 8; ++i) acc[i] = (v8f){0.f,0.f,0.f,0.f,0.f,0.f,0.f,0.f};
  float mrow[8], lrow[8];
  #pragma unroll
  for (int i = 0; i < 8; ++i) { mrow[i] = -1e30f; lrow[i] = 0.f; }

  const int ntiles = (qt * QT + QT) / KT;

  // ---- prologue: async-stage tile 0 into buffer 0 ----
  {
    #pragma unroll
    for (int i = 0; i < 4; ++i) {                 // K tile: 64x128
      int c = tid + i * 256, row = c >> 4, col8 = (c & 15) * 8;
      async_b128((unsigned)(uintptr_t)(kbuf + row * KSTR + col8),
                 kbase + (size_t)row * (HH * DD) + col8);
    }
    #pragma unroll
    for (int i = 0; i < 4; ++i) {                 // V^T tile: 128x64
      int c = tid + i * 256, row = c >> 3, col8 = (c & 7) * 8;
      async_b128((unsigned)(uintptr_t)(vbuf + row * VSTR + col8),
                 vTbase + (size_t)row * SS + col8);
    }
  }

  for (int t = 0; t < ntiles; ++t) {
    const int kv0 = t * KT;
    const int cur = t & 1;
    __bf16* kb = kbuf + cur * (KT * KSTR);
    __bf16* vb = vbuf + cur * (DD * VSTR);

    // ---- issue next tile into other buffer, then wait for current ----
    if (t + 1 < ntiles) {
      const int nxt = cur ^ 1;
      const int kv1 = (t + 1) * KT;
      __bf16* kn = kbuf + nxt * (KT * KSTR);
      __bf16* vn = vbuf + nxt * (DD * VSTR);
      #pragma unroll
      for (int i = 0; i < 4; ++i) {
        int c = tid + i * 256, row = c >> 4, col8 = (c & 15) * 8;
        async_b128((unsigned)(uintptr_t)(kn + row * KSTR + col8),
                   kbase + (size_t)(kv1 + row) * (HH * DD) + col8);
      }
      #pragma unroll
      for (int i = 0; i < 4; ++i) {
        int c = tid + i * 256, row = c >> 3, col8 = (c & 7) * 8;
        async_b128((unsigned)(uintptr_t)(vn + row * VSTR + col8),
                   vTbase + (size_t)row * SS + kv1 + col8);
      }
      asm volatile("s_wait_asynccnt 0x8" ::: "memory");  // in-order: tile t landed
    } else {
      asm volatile("s_wait_asynccnt 0x0" ::: "memory");
    }
    __syncthreads();

    if (kv0 <= qbase + 15) {
      // ---- S = Q K^T, 4 subtiles of 16 kv cols ----
      v8f sc[4];
      #pragma unroll
      for (int ns = 0; ns < 4; ++ns) {
        v16bf bk[4];
        #pragma unroll
        for (int ck = 0; ck < 4; ++ck)
          #pragma unroll
          for (int e = 0; e < 16; ++e)
            bk[ck][e] = kb[(ns * 16 + lh) * KSTR + ck * 32 + frag_k(e, hi)];
        v8f c = (v8f){0.f,0.f,0.f,0.f,0.f,0.f,0.f,0.f};
        #pragma unroll
        for (int ck = 0; ck < 4; ++ck)
          c = __builtin_amdgcn_wmma_f32_16x16x32_bf16(false, aq[ck], false, bk[ck],
                                                      (short)0, c, false, false);
        sc[ns] = c;
      }

      // ---- scale + causal mask ----
      #pragma unroll
      for (int ns = 0; ns < 4; ++ns) {
        int col = kv0 + ns * 16 + lh;
        #pragma unroll
        for (int r = 0; r < 8; ++r) {
          int qidx = qbase + r + (hi ? 8 : 0);
          sc[ns][r] = sc[ns][r] * scale + (col > qidx ? -1e9f : 0.f);
        }
      }

      // ---- online softmax ----
      #pragma unroll
      for (int r = 0; r < 8; ++r) {
        float mx = fmaxf(fmaxf(sc[0][r], sc[1][r]), fmaxf(sc[2][r], sc[3][r]));
        mx = fmaxf(mx, __shfl_xor(mx, 1));
        mx = fmaxf(mx, __shfl_xor(mx, 2));
        mx = fmaxf(mx, __shfl_xor(mx, 4));
        mx = fmaxf(mx, __shfl_xor(mx, 8));
        float mnew  = fmaxf(mrow[r], mx);
        float alpha = __expf(mrow[r] - mnew);
        float rs = 0.f;
        #pragma unroll
        for (int ns = 0; ns < 4; ++ns) {
          float p = __expf(sc[ns][r] - mnew);
          sc[ns][r] = p;
          rs += p;
        }
        rs += __shfl_xor(rs, 1);
        rs += __shfl_xor(rs, 2);
        rs += __shfl_xor(rs, 4);
        rs += __shfl_xor(rs, 8);
        lrow[r] = lrow[r] * alpha + rs;
        mrow[r] = mnew;
        #pragma unroll
        for (int dch = 0; dch < 8; ++dch) acc[dch][r] *= alpha;
      }

      // ---- stash P (C-layout -> row-major bf16 in per-wave LDS) ----
      __bf16* pw = pbuf + wave * 16 * PSTR;
      #pragma unroll
      for (int ns = 0; ns < 4; ++ns)
        #pragma unroll
        for (int r = 0; r < 8; ++r)
          pw[(r + (hi ? 8 : 0)) * PSTR + ns * 16 + lh] = f2bf(sc[ns][r]);
      asm volatile("s_wait_dscnt 0x0" ::: "memory");   // wave-private RAW

      // ---- O += P V ----
      v16bf ap[2];
      #pragma unroll
      for (int ck = 0; ck < 2; ++ck)
        #pragma unroll
        for (int e = 0; e < 16; ++e)
          ap[ck][e] = pw[lh * PSTR + ck * 32 + frag_k(e, hi)];
      #pragma unroll
      for (int dch = 0; dch < 8; ++dch) {
        v16bf bv0, bv1;
        #pragma unroll
        for (int e = 0; e < 16; ++e) {
          bv0[e] = vb[(dch * 16 + lh) * VSTR +      frag_k(e, hi)];
          bv1[e] = vb[(dch * 16 + lh) * VSTR + 32 + frag_k(e, hi)];
        }
        acc[dch] = __builtin_amdgcn_wmma_f32_16x16x32_bf16(false, ap[0], false, bv0,
                                                           (short)0, acc[dch], false, false);
        acc[dch] = __builtin_amdgcn_wmma_f32_16x16x32_bf16(false, ap[1], false, bv1,
                                                           (short)0, acc[dch], false, false);
      }
    }
    __syncthreads();   // frees buf[cur] for the issue at top of next iteration
  }

  // ---- epilogue ----
  #pragma unroll
  for (int r = 0; r < 8; ++r) {
    int qidx = qbase + r + (hi ? 8 : 0);
    float inv = 1.f / lrow[r];
    float* orow = out + (((size_t)b * SS + qidx) * HH + h) * DD;
    #pragma unroll
    for (int dch = 0; dch < 8; ++dch)
      orow[dch * 16 + lh] = acc[dch][r] * inv;
  }
}

// ---------------- fallback: direct-fp32 kernel (round-1 version) ----------------
__global__ __launch_bounds__(256, 1)
void fa_fwd_f32_kernel(const float* __restrict__ xq, const float* __restrict__ xk,
                       const float* __restrict__ xv, float* __restrict__ out) {
  __shared__ __bf16 kt[KT][DD + 8];
  __shared__ __bf16 vt[KT][DD + 8];
  __shared__ __bf16 pst[NW][16][KT + 8];

  const int tid  = threadIdx.x;
  const int wave = tid >> 5;
  const int lane = tid & 31;
  const int lh   = lane & 15;
  const int hi   = (lane >> 4) & 1;

  const int qtiles = SS / QT;
  const int qt = blockIdx.x % qtiles;
  const int h  = (blockIdx.x / qtiles) % HH;
  const int b  = blockIdx.x / (qtiles * HH);
  const int qbase = qt * QT + wave * 16;
  const float scale = 0.08838834764831845f;

  v16bf aq[4];
  {
    const float* qrow = xq + (((size_t)b * SS + (qbase + lh)) * HH + h) * DD;
    #pragma unroll
    for (int c = 0; c < 4; ++c)
      #pragma unroll
      for (int e = 0; e < 16; ++e)
        aq[c][e] = f2bf(qrow[c * 32 + frag_k(e, hi)]);
  }

  v8f acc[8];
  #pragma unroll
  for (int i = 0; i < 8; ++i) acc[i] = (v8f){0.f,0.f,0.f,0.f,0.f,0.f,0.f,0.f};
  float mrow[8], lrow[8];
  #pragma unroll
  for (int i = 0; i < 8; ++i) { mrow[i] = -1e30f; lrow[i] = 0.f; }

  const int ntiles = (qt * QT + QT) / KT;
  for (int t = 0; t < ntiles; ++t) {
    const int kv0 = t * KT;
    for (int i = tid; i < KT * (DD / 4); i += 256) {
      int r = i / (DD / 4), c4 = (i % (DD / 4)) * 4;
      size_t base = (((size_t)b * SS + (kv0 + r)) * HH + h) * DD + c4;
      const float4 k4 = *(const float4*)(xk + base);
      const float4 v4 = *(const float4*)(xv + base);
      kt[r][c4+0] = f2bf(k4.x); kt[r][c4+1] = f2bf(k4.y);
      kt[r][c4+2] = f2bf(k4.z); kt[r][c4+3] = f2bf(k4.w);
      vt[r][c4+0] = f2bf(v4.x); vt[r][c4+1] = f2bf(v4.y);
      vt[r][c4+2] = f2bf(v4.z); vt[r][c4+3] = f2bf(v4.w);
    }
    __syncthreads();

    if (kv0 <= qbase + 15) {
      v8f sc[4];
      #pragma unroll
      for (int ns = 0; ns < 4; ++ns) {
        v8f c = (v8f){0.f,0.f,0.f,0.f,0.f,0.f,0.f,0.f};
        #pragma unroll
        for (int ck = 0; ck < 4; ++ck) {
          v16bf bk;
          #pragma unroll
          for (int e = 0; e < 16; ++e)
            bk[e] = kt[ns * 16 + lh][ck * 32 + frag_k(e, hi)];
          c = __builtin_amdgcn_wmma_f32_16x16x32_bf16(false, aq[ck], false, bk,
                                                      (short)0, c, false, false);
        }
        sc[ns] = c;
      }
      #pragma unroll
      for (int ns = 0; ns < 4; ++ns) {
        int col = kv0 + ns * 16 + lh;
        #pragma unroll
        for (int r = 0; r < 8; ++r) {
          int qidx = qbase + r + (hi ? 8 : 0);
          sc[ns][r] = sc[ns][r] * scale + (col > qidx ? -1e9f : 0.f);
        }
      }
      #pragma unroll
      for (int r = 0; r < 8; ++r) {
        float mx = fmaxf(fmaxf(sc[0][r], sc[1][r]), fmaxf(sc[2][r], sc[3][r]));
        mx = fmaxf(mx, __shfl_xor(mx, 1));
        mx = fmaxf(mx, __shfl_xor(mx, 2));
        mx = fmaxf(mx, __shfl_xor(mx, 4));
        mx = fmaxf(mx, __shfl_xor(mx, 8));
        float mnew  = fmaxf(mrow[r], mx);
        float alpha = __expf(mrow[r] - mnew);
        float rs = 0.f;
        #pragma unroll
        for (int ns = 0; ns < 4; ++ns) {
          float p = __expf(sc[ns][r] - mnew);
          sc[ns][r] = p; rs += p;
        }
        rs += __shfl_xor(rs, 1); rs += __shfl_xor(rs, 2);
        rs += __shfl_xor(rs, 4); rs += __shfl_xor(rs, 8);
        lrow[r] = lrow[r] * alpha + rs;
        mrow[r] = mnew;
        #pragma unroll
        for (int dch = 0; dch < 8; ++dch) acc[dch][r] *= alpha;
      }
      #pragma unroll
      for (int ns = 0; ns < 4; ++ns)
        #pragma unroll
        for (int r = 0; r < 8; ++r)
          pst[wave][r + (hi ? 8 : 0)][ns * 16 + lh] = f2bf(sc[ns][r]);
      asm volatile("s_wait_dscnt 0x0" ::: "memory");
      #pragma unroll
      for (int ck = 0; ck < 2; ++ck) {
        v16bf ap;
        #pragma unroll
        for (int e = 0; e < 16; ++e)
          ap[e] = pst[wave][lh][ck * 32 + frag_k(e, hi)];
        #pragma unroll
        for (int dch = 0; dch < 8; ++dch) {
          v16bf bv;
          #pragma unroll
          for (int e = 0; e < 16; ++e)
            bv[e] = vt[ck * 32 + frag_k(e, hi)][dch * 16 + lh];
          acc[dch] = __builtin_amdgcn_wmma_f32_16x16x32_bf16(false, ap, false, bv,
                                                             (short)0, acc[dch], false, false);
        }
      }
    }
    __syncthreads();
  }

  #pragma unroll
  for (int r = 0; r < 8; ++r) {
    int qidx = qbase + r + (hi ? 8 : 0);
    float inv = 1.f / lrow[r];
    float* orow = out + (((size_t)b * SS + qidx) * HH + h) * DD;
    #pragma unroll
    for (int dch = 0; dch < 8; ++dch)
      orow[dch * 16 + lh] = acc[dch][r] * inv;
  }
}

extern "C" void kernel_launch(void* const* d_in, const int* in_sizes, int n_in,
                              void* d_out, int out_size, void* d_ws, size_t ws_size,
                              hipStream_t stream) {
  const float* xq = (const float*)d_in[0];
  const float* xk = (const float*)d_in[1];
  const float* xv = (const float*)d_in[2];
  float* out = (float*)d_out;

  const size_t elems = (size_t)BB * SS * HH * DD;   // 16,777,216
  const size_t need  = elems * 2 * 3;               // 3 bf16 tensors
  dim3 fgrid(BB * HH * (SS / QT));                  // 1024
  dim3 fblk(256);

  if (ws_size >= need) {
    __bf16* wq  = (__bf16*)d_ws;
    __bf16* wk  = wq + elems;
    __bf16* wvT = wk + elems;
    int n4 = (int)(elems / 4);
    dim3 cgrid((n4 + 255) / 256), cblk(256);
    cvt_bf16_kernel<<<cgrid, cblk, 0, stream>>>(xq, wq, n4);
    cvt_bf16_kernel<<<cgrid, cblk, 0, stream>>>(xk, wk, n4);
    cvtT_bf16_kernel<<<cgrid, cblk, 0, stream>>>(xv, wvT);
    fa_fwd_bf16_kernel<<<fgrid, fblk, SMEM_BYTES, stream>>>(wq, wk, wvT, out);
  } else {
    fa_fwd_f32_kernel<<<fgrid, fblk, 0, stream>>>(xq, xk, xv, out);
  }
}
